// GAT_15831249453669
// MI455X (gfx1250) — matrix operand order repaired and checked by reading, not compiled, not run on hardware
//
#include <hip/hip_runtime.h>
#include <hip/hip_bf16.h>

// ---------------- CDNA5 WMMA types ----------------
typedef __attribute__((ext_vector_type(16))) __bf16 v16bf;
typedef __attribute__((ext_vector_type(8)))  float  v8f;

#define GAT_H 3

// ---------------- helpers ----------------
__device__ __forceinline__ float elu1(float x) {
    return x > 0.f ? x : (__expf(x) - 1.f);
}

__device__ __forceinline__ void atomicMaxFloat(float* addr, float val) {
    // ordered-int trick; buffer initialized to -inf
    if (val >= 0.f)
        atomicMax((int*)addr, __float_as_int(val));
    else
        atomicMin((unsigned int*)addr, __float_as_uint(val));
}

// ---------------- small utility kernels ----------------
__global__ void fill_f32(float* __restrict__ p, float v, size_t n) {
    size_t i = (size_t)blockIdx.x * blockDim.x + threadIdx.x;
    if (i < n) p[i] = v;
}

// ---------------- pre-swizzle A: f32 [N,Fin] -> bf16 fragment panels ----------------
// Panel layout: [kChunk][mPanel][lane][16] bf16 (1 KB per 16x32 tile).
// A-fragment map (16-bit A 16x32): lane = (K%16)/8*16 + row%16, elem i = (K/16)*8 + K%8
__global__ void swizzle_A(const float* __restrict__ X, __bf16* __restrict__ Apan,
                          int Nrows, int Fin, int Mpad) {
    int idx = blockIdx.x * 256 + threadIdx.x;
    if (idx >= Mpad * Fin) return;
    int row = idx / Fin;
    int k   = idx - row * Fin;
    float v = (row < Nrows) ? X[(size_t)row * Fin + k] : 0.f;
    int p = row >> 4, r = row & 15;
    int kc = k >> 5, kk = k & 31;
    int half = (kk & 15) >> 3;
    int i    = ((kk >> 4) << 3) | (kk & 7);
    int lane = (half << 4) | r;
    int nPanM = Mpad >> 4;
    Apan[(((size_t)kc * nPanM + p) << 9) + (lane << 4) + i] = (__bf16)v;
}

// ---------------- pre-swizzle B: f32 [Fin,Fout] -> bf16 fragment panels ----------------
// Panel layout: [kChunk][nPanel][lane][16] bf16.
// B-fragment map (32x16): lane = (K%16)/8*16 + col%16, elem i = (K/16)*8 + K%8
__global__ void swizzle_B(const float* __restrict__ W, __bf16* __restrict__ Bpan,
                          int Fin, int Fout, int Npad) {
    int idx = blockIdx.x * 256 + threadIdx.x;
    if (idx >= Fin * Npad) return;
    int k   = idx / Npad;
    int col = idx - k * Npad;
    float v = (col < Fout) ? W[(size_t)k * Fout + col] : 0.f;
    int kc = k >> 5, kk = k & 31;
    int lane = (((kk & 15) >> 3) << 4) | (col & 15);
    int i    = ((kk >> 4) << 3) | (kk & 7);
    int p    = col >> 4;
    int nPanN = Npad >> 4;
    Bpan[(((size_t)kc * nPanN + p) << 9) + (lane << 4) + i] = (__bf16)v;
}

// ---------------- WMMA GEMM on pre-swizzled panels ----------------
// Block tile 128x128, BK=32, 256 threads = 8 waves (4 along M, 2 along N),
// wave tile 32x64 = 2x4 accumulator tiles of 16x16 f32.
// Staging: contiguous 8KB A + 8KB B per k-chunk via global_load_async_to_lds_b128,
// double-buffered in LDS, pipelined with s_wait_asynccnt.
__global__ __launch_bounds__(256) void gat_gemm_wmma(
    const __bf16* __restrict__ Apan, const __bf16* __restrict__ Bpan,
    float* __restrict__ Hout, int Nrows, int Fout,
    int kChunks, int nPanM, int nPanN)
{
    __shared__ __attribute__((aligned(16))) __bf16 sA[2][4096];
    __shared__ __attribute__((aligned(16))) __bf16 sB[2][4096];

    const int tid   = threadIdx.x;
    const int lane  = tid & 31;
    const int wave  = tid >> 5;
    const int waveM = wave >> 1;   // 0..3
    const int waveN = wave & 1;    // 0..1
    const int pm = blockIdx.x * 8; // A panel base
    const int pn = blockIdx.y * 8; // B panel base

    const unsigned ldsA = (unsigned)(uintptr_t)&sA[0][0];
    const unsigned ldsB = (unsigned)(uintptr_t)&sB[0][0];

    v8f acc[2][4];
#pragma unroll
    for (int i = 0; i < 2; ++i)
#pragma unroll
        for (int j = 0; j < 4; ++j)
            acc[i][j] = (v8f){0.f, 0.f, 0.f, 0.f, 0.f, 0.f, 0.f, 0.f};

    // async-copy one k-chunk (8KB A + 8KB B) into LDS buffer `buf`
    auto issue_stage = [&](int kc, int buf) {
        uint64_t ga = (uint64_t)(uintptr_t)(Apan + (((size_t)kc * nPanM + pm) << 9));
        uint64_t gb = (uint64_t)(uintptr_t)(Bpan + (((size_t)kc * nPanN + pn) << 9));
        unsigned lA = ldsA + (unsigned)buf * 8192u;
        unsigned lB = ldsB + (unsigned)buf * 8192u;
#pragma unroll
        for (int it = 0; it < 2; ++it) {
            unsigned boff = (unsigned)(tid + it * 256) * 16u;   // 0..8176
            asm volatile("global_load_async_to_lds_b128 %0, %1, off"
                         :: "v"(lA + boff), "v"(ga + boff) : "memory");
            asm volatile("global_load_async_to_lds_b128 %0, %1, off"
                         :: "v"(lB + boff), "v"(gb + boff) : "memory");
        }
    };

    issue_stage(0, 0);
    for (int kc = 0; kc < kChunks; ++kc) {
        const int cur = kc & 1;
        if (kc + 1 < kChunks) {
            issue_stage(kc + 1, cur ^ 1);
            // 4 newly-issued outstanding allowed; current chunk's 4 must be done
            asm volatile("s_wait_asynccnt 0x4" ::: "memory");
        } else {
            asm volatile("s_wait_asynccnt 0x0" ::: "memory");
        }
        __syncthreads();

        v16bf a0 = *(const v16bf*)&sA[cur][(waveM * 2 + 0) * 512 + lane * 16];
        v16bf a1 = *(const v16bf*)&sA[cur][(waveM * 2 + 1) * 512 + lane * 16];
#pragma unroll
        for (int j = 0; j < 4; ++j) {
            v16bf b = *(const v16bf*)&sB[cur][(waveN * 4 + j) * 512 + lane * 16];
            acc[0][j] = __builtin_amdgcn_wmma_f32_16x16x32_bf16(
                false, a0, false, b, (short)0, acc[0][j], false, false);
            acc[1][j] = __builtin_amdgcn_wmma_f32_16x16x32_bf16(
                false, a1, false, b, (short)0, acc[1][j], false, false);
        }
        __syncthreads();
    }

    // ---- store: C/D layout: VGPR r -> M = r + 8*(lane/16), N = lane%16 ----
    const int half = lane >> 4;
    const int cn   = lane & 15;
#pragma unroll
    for (int mi = 0; mi < 2; ++mi)
#pragma unroll
        for (int nj = 0; nj < 4; ++nj)
#pragma unroll
            for (int r = 0; r < 8; ++r) {
                int row = blockIdx.x * 128 + (waveM * 2 + mi) * 16 + r + half * 8;
                int col = blockIdx.y * 128 + (waveN * 4 + nj) * 16 + cn;
                if (row < Nrows && col < Fout)
                    Hout[(size_t)row * Fout + col] = acc[mi][nj][r];
            }
}

// ---------------- alpha_src / alpha_dst: per-node per-head dot products ----------------
__global__ __launch_bounds__(256) void gat_alpha(
    const float* __restrict__ Hf, const float* __restrict__ att_s,
    const float* __restrict__ att_d, float* __restrict__ asrc,
    float* __restrict__ adst, int Nrows, int C)
{
    int lane = threadIdx.x & 31;
    int node = blockIdx.x * 8 + (threadIdx.x >> 5);
    if (node >= Nrows) return;
#pragma unroll
    for (int h = 0; h < GAT_H; ++h) {
        float ss = 0.f, sd = 0.f;
        const float* hp = Hf + (size_t)node * (GAT_H * C) + h * C;
        for (int c = lane; c < C; c += 32) {
            float v = hp[c];
            ss += v * att_s[h * C + c];
            sd += v * att_d[h * C + c];
        }
#pragma unroll
        for (int o = 16; o > 0; o >>= 1) {
            ss += __shfl_down(ss, o, 32);
            sd += __shfl_down(sd, o, 32);
        }
        if (lane == 0) {
            asrc[node * GAT_H + h] = ss;
            adst[node * GAT_H + h] = sd;
        }
    }
}

// ---------------- edge pass 1: segment max of leaky_relu(e) ----------------
__global__ void gat_edge_max(const int* __restrict__ ei, int E, int Nrows,
                             const float* __restrict__ asrc,
                             const float* __restrict__ adst,
                             float* __restrict__ m)
{
    int idx = blockIdx.x * blockDim.x + threadIdx.x;
    int ET = E + Nrows;
    if (idx >= ET * GAT_H) return;
    int e = idx / GAT_H, h = idx - e * GAT_H;
    int src = (e < E) ? ei[e]     : (e - E);
    int dst = (e < E) ? ei[E + e] : (e - E);
    float v = asrc[src * GAT_H + h] + adst[dst * GAT_H + h];
    v = v > 0.f ? v : 0.2f * v;
    atomicMaxFloat(&m[dst * GAT_H + h], v);
}

// ---------------- edge pass 2: ex = exp(e - m[dst]); den += ex ----------------
__global__ void gat_edge_exp(const int* __restrict__ ei, int E, int Nrows,
                             const float* __restrict__ asrc,
                             const float* __restrict__ adst,
                             const float* __restrict__ m,
                             float* __restrict__ exb, float* __restrict__ den)
{
    int idx = blockIdx.x * blockDim.x + threadIdx.x;
    int ET = E + Nrows;
    if (idx >= ET * GAT_H) return;
    int e = idx / GAT_H, h = idx - e * GAT_H;
    int src = (e < E) ? ei[e]     : (e - E);
    int dst = (e < E) ? ei[E + e] : (e - E);
    float v = asrc[src * GAT_H + h] + adst[dst * GAT_H + h];
    v = v > 0.f ? v : 0.2f * v;
    float ex = __expf(v - m[dst * GAT_H + h]);
    exb[idx] = ex;
    atomicAdd(&den[dst * GAT_H + h], ex);
}

// ---------------- edge pass 3: agg[dst] += (ex/den) * h[src] (wave per edge-head) ----------------
__global__ __launch_bounds__(256) void gat_edge_agg(
    const int* __restrict__ ei, int E, int Nrows, int C,
    const float* __restrict__ Hf, const float* __restrict__ exb,
    const float* __restrict__ den, float* __restrict__ agg)
{
    int lane = threadIdx.x & 31;
    int gw   = (blockIdx.x * 256 + threadIdx.x) >> 5;
    int ET = E + Nrows;
    if (gw >= ET * GAT_H) return;
    int e = gw / GAT_H, h = gw - e * GAT_H;
    int src = (e < E) ? ei[e]     : (e - E);
    int dst = (e < E) ? ei[E + e] : (e - E);
    float w = exb[gw] / (den[dst * GAT_H + h] + 1e-16f);
    const float* hp = Hf  + (size_t)src * (GAT_H * C) + h * C;
    float*       op = agg + (size_t)dst * (GAT_H * C) + h * C;
    for (int c = lane; c < C; c += 32)
        atomicAdd(&op[c], w * hp[c]);
}

// ---------------- finalize: elu(agg + bias) in-place (concat layers) ----------------
__global__ void gat_finalize_concat(float* __restrict__ buf,
                                    const float* __restrict__ bias,
                                    int Nrows, int Fout)
{
    int idx = blockIdx.x * blockDim.x + threadIdx.x;
    if (idx >= Nrows * Fout) return;
    float v = buf[idx] + bias[idx % Fout];
    buf[idx] = elu1(v);
}

// ---------------- finalize: mean over heads + bias + elu -> out ----------------
__global__ void gat_finalize_mean(const float* __restrict__ agg,
                                  const float* __restrict__ bias,
                                  float* __restrict__ out, int Nrows, int Oc)
{
    int idx = blockIdx.x * blockDim.x + threadIdx.x;
    if (idx >= Nrows * Oc) return;
    int n = idx / Oc, c = idx - n * Oc;
    const float* p = agg + (size_t)n * (GAT_H * Oc);
    float s = (p[c] + p[Oc + c] + p[2 * Oc + c]) * (1.f / 3.f);
    out[idx] = elu1(s + bias[c]);
}

// ---------------- launcher ----------------
extern "C" void kernel_launch(void* const* d_in, const int* in_sizes, int n_in,
                              void* d_out, int out_size, void* d_ws, size_t ws_size,
                              hipStream_t stream) {
    (void)n_in; (void)out_size; (void)ws_size;
    const float* x   = (const float*)d_in[0];
    const int*   ei  = (const int*)d_in[1];
    const float* W1  = (const float*)d_in[2];
    const float* as1 = (const float*)d_in[3];
    const float* ad1 = (const float*)d_in[4];
    const float* b1  = (const float*)d_in[5];
    const float* W2  = (const float*)d_in[6];
    const float* as2 = (const float*)d_in[7];
    const float* ad2 = (const float*)d_in[8];
    const float* b2  = (const float*)d_in[9];
    const float* W3  = (const float*)d_in[10];
    const float* as3 = (const float*)d_in[11];
    const float* ad3 = (const float*)d_in[12];
    const float* b3  = (const float*)d_in[13];

    const int N  = in_sizes[0] / 128;  // 30000
    const int E  = in_sizes[1] / 2;    // 480000
    const int ET = E + N;              // with self loops
    const int Mpad = ((N + 127) / 128) * 128;

    // ---- carve workspace ----
    char* ws = (char*)d_ws;
    size_t off = 0;
    auto carve = [&](size_t bytes) -> void* {
        void* p = ws + off;
        off = (off + bytes + 255) & ~(size_t)255;
        return p;
    };
    float*  bufA  = (float*)carve((size_t)N * 768 * sizeof(float)); // GEMM out h
    float*  bufB  = (float*)carve((size_t)N * 768 * sizeof(float)); // agg / layer out
    __bf16* Abf   = (__bf16*)carve((size_t)Mpad * 768 * sizeof(__bf16)); // A panels (max Fin=768)
    __bf16* Bbf   = (__bf16*)carve((size_t)768 * 768 * sizeof(__bf16));  // B panels (max)
    float*  asrc  = (float*)carve((size_t)N * GAT_H * sizeof(float));
    float*  adst  = (float*)carve((size_t)N * GAT_H * sizeof(float));
    float*  mbuf  = (float*)carve((size_t)N * GAT_H * sizeof(float));
    float*  dbuf  = (float*)carve((size_t)N * GAT_H * sizeof(float));
    float*  exbuf = (float*)carve((size_t)ET * GAT_H * sizeof(float));
    float*  bufC  = (float*)carve((size_t)N * 96 * sizeof(float));  // layer-3 agg

    const float NEG_INF = -__builtin_inff();

    auto run_layer = [&](const float* Xin, const float* W, const float* as,
                         const float* ad, int Fin, int Fout, int C,
                         float* Hbuf, float* Abuf) {
        const int kChunks = Fin / 32;
        const int nPanM   = Mpad / 16;
        const int Npad    = ((Fout + 127) / 128) * 128;
        const int nPanN   = Npad / 16;

        int nA = Mpad * Fin;
        swizzle_A<<<(nA + 255) / 256, 256, 0, stream>>>(Xin, Abf, N, Fin, Mpad);
        int nB = Fin * Npad;
        swizzle_B<<<(nB + 255) / 256, 256, 0, stream>>>(W, Bbf, Fin, Fout, Npad);

        dim3 g(Mpad / 128, Npad / 128);
        gat_gemm_wmma<<<g, 256, 0, stream>>>(Abf, Bbf, Hbuf, N, Fout,
                                             kChunks, nPanM, nPanN);

        gat_alpha<<<(N + 7) / 8, 256, 0, stream>>>(Hbuf, as, ad, asrc, adst, N, C);
        size_t nh = (size_t)N * GAT_H;
        fill_f32<<<(unsigned)((nh + 255) / 256), 256, 0, stream>>>(mbuf, NEG_INF, nh);
        fill_f32<<<(unsigned)((nh + 255) / 256), 256, 0, stream>>>(dbuf, 0.f, nh);
        int nEH = ET * GAT_H;
        gat_edge_max<<<(nEH + 255) / 256, 256, 0, stream>>>(ei, E, N, asrc, adst, mbuf);
        gat_edge_exp<<<(nEH + 255) / 256, 256, 0, stream>>>(ei, E, N, asrc, adst, mbuf, exbuf, dbuf);
        size_t nOut = (size_t)N * Fout;
        fill_f32<<<(unsigned)((nOut + 255) / 256), 256, 0, stream>>>(Abuf, 0.f, nOut);
        int aggBlocks = (nEH * 32 + 255) / 256;
        gat_edge_agg<<<aggBlocks, 256, 0, stream>>>(ei, E, N, C, Hbuf, exbuf, dbuf, Abuf);
    };

    // Layer 1: x[N,128] -> bufA (h1), agg -> bufB, elu+bias in place
    run_layer(x, W1, as1, ad1, 128, 768, 256, bufA, bufB);
    gat_finalize_concat<<<(N * 768 + 255) / 256, 256, 0, stream>>>(bufB, b1, N, 768);

    // Layer 2: bufB[N,768] -> bufA (h2), agg -> bufB (safe: GEMM consumed bufB first)
    run_layer(bufB, W2, as2, ad2, 768, 768, 256, bufA, bufB);
    gat_finalize_concat<<<(N * 768 + 255) / 256, 256, 0, stream>>>(bufB, b2, N, 768);

    // Layer 3: bufB[N,768] -> bufA[N,96], agg -> bufC, mean heads + bias + elu -> d_out
    run_layer(bufB, W3, as3, ad3, 768, 96, 32, bufA, bufC);
    gat_finalize_mean<<<(N * 32 + 255) / 256, 256, 0, stream>>>(bufC, b3, (float*)d_out, N, 32);
}
